// MLPTime_20779051778730
// MI455X (gfx1250) — matrix operand order repaired and checked by reading, not compiled, not run on hardware
//
#include <hip/hip_runtime.h>

typedef __attribute__((ext_vector_type(16))) _Float16 v16h;
typedef __attribute__((ext_vector_type(8)))  float    v8f;

#define N_TOK     16384        // 32*512 tokens
#define S_DIM     512
#define N_EXP     8
#define OUT_ELEMS (N_TOK * S_DIM)

// ---- workspace layout (bytes), all regions 256B aligned ----
#define WS_COUNTS 0                                   // 8 ints
#define WS_TOK    256                                 // 8*16384 int
#define WS_WT     (WS_TOK + N_EXP * N_TOK * 4)        // 8*16384 float
#define WS_WH     (WS_WT  + N_EXP * N_TOK * 4)        // 8*512*512 f16
#define WS_XH     (WS_WH  + N_EXP * S_DIM * S_DIM * 2)// 16384*512 f16
// total ~21.5 MB

union H4 { _Float16 h[4];  uint2 q; };
union HV { v16h v; uint4 q[2]; };

// ---------------- kernel 1: zero out-accumulator + counters ----------------
__global__ __launch_bounds__(256) void zero_kernel(float* __restrict__ out,
                                                   int* __restrict__ counts) {
    size_t i = (size_t)blockIdx.x * blockDim.x + threadIdx.x;
    float4 z = {0.f, 0.f, 0.f, 0.f};
    ((float4*)out)[i] = z;                      // grid sized exactly OUT_ELEMS/4
    if (blockIdx.x == 0 && threadIdx.x < N_EXP) counts[threadIdx.x] = 0;
}

// ---------------- kernel 2: expert_w fp32 -> f16 ----------------
__global__ __launch_bounds__(256) void convw_kernel(const float* __restrict__ w,
                                                    _Float16* __restrict__ wh) {
    size_t i = ((size_t)blockIdx.x * blockDim.x + threadIdx.x) * 4;
    float4 f = *(const float4*)(w + i);
    H4 h;
    h.h[0] = (_Float16)f.x; h.h[1] = (_Float16)f.y;
    h.h[2] = (_Float16)f.z; h.h[3] = (_Float16)f.w;
    *(uint2*)(wh + i) = h.q;
}

// ---------------- kernel 3: gating + softmax + top2 + x->f16 ----------------
__global__ __launch_bounds__(256) void gate_kernel(
    const float* __restrict__ x, const float* __restrict__ gate_w,
    float* __restrict__ gates_out, _Float16* __restrict__ xh,
    int* __restrict__ counts, int* __restrict__ etok, float* __restrict__ ewt) {
    __shared__ float gs[N_EXP * S_DIM];          // 16 KB gate weights
    const int tid = threadIdx.x;
    {
        const float4* src = (const float4*)gate_w;
        float4* dst = (float4*)gs;
        for (int i = tid; i < N_EXP * S_DIM / 4; i += 256) dst[i] = src[i];
    }
    __syncthreads();

    const int wave = tid >> 5, lane = tid & 31;
    const int tok  = blockIdx.x * 8 + wave;      // one wave32 per token
    const float* xr = x + (size_t)tok * S_DIM + lane * 16;

    float xv[16];
    {
        float4 f0 = ((const float4*)xr)[0], f1 = ((const float4*)xr)[1];
        float4 f2 = ((const float4*)xr)[2], f3 = ((const float4*)xr)[3];
        xv[0]=f0.x; xv[1]=f0.y; xv[2]=f0.z; xv[3]=f0.w;
        xv[4]=f1.x; xv[5]=f1.y; xv[6]=f1.z; xv[7]=f1.w;
        xv[8]=f2.x; xv[9]=f2.y; xv[10]=f2.z; xv[11]=f2.w;
        xv[12]=f3.x; xv[13]=f3.y; xv[14]=f3.z; xv[15]=f3.w;
    }
    // convert this token's slice to f16 while it's in registers
    {
        union { _Float16 h[16]; uint4 q[2]; } hv;
        #pragma unroll
        for (int q = 0; q < 16; q++) hv.h[q] = (_Float16)xv[q];
        uint4* dst = (uint4*)(xh + (size_t)tok * S_DIM + lane * 16);
        dst[0] = hv.q[0]; dst[1] = hv.q[1];
    }
    float acc[N_EXP];
    #pragma unroll
    for (int e = 0; e < N_EXP; e++) acc[e] = 0.f;
    #pragma unroll
    for (int e = 0; e < N_EXP; e++) {
        const float* g = gs + e * S_DIM + lane * 16;
        #pragma unroll
        for (int q = 0; q < 16; q++) acc[e] += xv[q] * g[q];
    }
    // wave32 reduction
    #pragma unroll
    for (int e = 0; e < N_EXP; e++)
        for (int off = 16; off; off >>= 1)
            acc[e] += __shfl_xor(acc[e], off, 32);

    if (lane == 0) {
        float mx = acc[0];
        #pragma unroll
        for (int e = 1; e < N_EXP; e++) mx = fmaxf(mx, acc[e]);
        float g[N_EXP], s = 0.f;
        #pragma unroll
        for (int e = 0; e < N_EXP; e++) { g[e] = __expf(acc[e] - mx); s += g[e]; }
        float inv = 1.f / s;
        #pragma unroll
        for (int e = 0; e < N_EXP; e++) {
            g[e] *= inv;
            gates_out[(size_t)tok * N_EXP + e] = g[e];
        }
        // top-2 (ties -> lowest index, matching lax.top_k)
        int e0 = 0; float w0 = g[0];
        #pragma unroll
        for (int e = 1; e < N_EXP; e++) if (g[e] > w0) { w0 = g[e]; e0 = e; }
        int e1 = -1; float w1 = -1.f;
        #pragma unroll
        for (int e = 0; e < N_EXP; e++)
            if (e != e0 && g[e] > w1) { w1 = g[e]; e1 = e; }
        int s0 = atomicAdd(counts + e0, 1);
        etok[e0 * N_TOK + s0] = tok; ewt[e0 * N_TOK + s0] = w0;
        int s1 = atomicAdd(counts + e1, 1);
        etok[e1 * N_TOK + s1] = tok; ewt[e1 * N_TOK + s1] = w1;
    }
}

// ---------------- kernel 4: compacted grouped GEMM (WMMA f16) ----------------
// Tile: 32 rows x 256 cols per block; 8 waves, each wave = 2 row-blocks x 2
// col-subtiles = 4 x v_wmma_f32_16x16x32_f16 per K-step (B reused 2x, A 2x).
// grid: (N_TOK/32 m-tiles, S_DIM/256 n-tiles, N_EXP experts); block 256
__global__ __launch_bounds__(256) void moe_gemm_kernel(
    const _Float16* __restrict__ xh, const _Float16* __restrict__ wh,
    const float* __restrict__ expert_b, const int* __restrict__ counts,
    const int* __restrict__ etok, const float* __restrict__ ewt,
    float* __restrict__ out) {
    const int e   = blockIdx.z;
    const int cnt = counts[e];
    const int m0  = blockIdx.x * 32;
    if (m0 >= cnt) return;

    __shared__ __align__(16) _Float16 As[32 * S_DIM];   // 32 gathered rows, 32 KB
    __shared__ int   toks[32];
    __shared__ float wts[32];

    const int tid = threadIdx.x;
    if (tid < 32) {
        int slot = m0 + tid;
        int t = -1; float w = 0.f;
        if (slot < cnt) { t = etok[e * N_TOK + slot]; w = ewt[e * N_TOK + slot]; }
        toks[tid] = t; wts[tid] = w;
    }
    __syncthreads();
    {   // stage A: 32 rows x 512 f16; 8 threads/row, 128B each
        int r = tid >> 3, seg = tid & 7;
        uint4* dst = (uint4*)(As + r * S_DIM + seg * 64);
        int t = toks[r];
        if (t >= 0) {
            const uint4* src = (const uint4*)(xh + (size_t)t * S_DIM + seg * 64);
            #pragma unroll
            for (int q = 0; q < 8; q++) dst[q] = src[q];
        } else {
            uint4 z = {0u, 0u, 0u, 0u};
            #pragma unroll
            for (int q = 0; q < 8; q++) dst[q] = z;
        }
    }
    __syncthreads();

    const int wave  = tid >> 5, lane = tid & 31;
    const int khalf = lane >> 4, nl = lane & 15;
    const int col0  = blockIdx.y * 256 + wave * 32;      // 2 x 16-col subtiles/wave
    // B rows are K-contiguous in wh[e][o][i] -> matches B-fragment layout directly
    const _Float16* b0p = wh + ((size_t)e * S_DIM + col0 + nl) * S_DIM;
    const _Float16* b1p = b0p + (size_t)16 * S_DIM;
    // A fragments: lane = row m (=nl), elements 0..15: K=16*khalf+0..15
    const _Float16* a0p = As + (size_t)nl * S_DIM + 16 * khalf;         // rows 0..15
    const _Float16* a1p = a0p + (size_t)16 * S_DIM;                     // rows 16..31

    v8f acc00 = {}; v8f acc01 = {};   // rows 0..15  x cols {sub0, sub1}
    v8f acc10 = {}; v8f acc11 = {};   // rows 16..31 x cols {sub0, sub1}
    #pragma unroll 2
    for (int kb = 0; kb < S_DIM; kb += 32) {
        HV a0, a1, b0, b1;
        a0.q[0] = *(const uint4*)(a0p + kb);
        a0.q[1] = *(const uint4*)(a0p + kb + 8);
        a1.q[0] = *(const uint4*)(a1p + kb);
        a1.q[1] = *(const uint4*)(a1p + kb + 8);
        const uint4* bq0 = (const uint4*)(b0p + kb + 16 * khalf);
        b0.q[0] = bq0[0]; b0.q[1] = bq0[1];
        const uint4* bq1 = (const uint4*)(b1p + kb + 16 * khalf);
        b1.q[0] = bq1[0]; b1.q[1] = bq1[1];
        __builtin_prefetch(b0p + kb + 128, 0, 0);
        acc00 = __builtin_amdgcn_wmma_f32_16x16x32_f16(false, a0.v, false, b0.v,
                                                       (short)0, acc00, false, false);
        acc01 = __builtin_amdgcn_wmma_f32_16x16x32_f16(false, a0.v, false, b1.v,
                                                       (short)0, acc01, false, false);
        acc10 = __builtin_amdgcn_wmma_f32_16x16x32_f16(false, a1.v, false, b0.v,
                                                       (short)0, acc10, false, false);
        acc11 = __builtin_amdgcn_wmma_f32_16x16x32_f16(false, a1.v, false, b1.v,
                                                       (short)0, acc11, false, false);
    }
    // epilogue: D element j -> m = j + 8*khalf (+16 for row-block 1), n = nl
    const int c0 = col0 + nl, c1 = c0 + 16;
    const float bias0 = expert_b[e * S_DIM + c0];
    const float bias1 = expert_b[e * S_DIM + c1];
    #pragma unroll
    for (int j = 0; j < 8; j++) {
        int m = j + 8 * khalf;
        int t = toks[m];
        if (t >= 0) {
            float w = wts[m];
            unsafeAtomicAdd(out + (size_t)t * S_DIM + c0, w * (acc00[j] + bias0));
            unsafeAtomicAdd(out + (size_t)t * S_DIM + c1, w * (acc01[j] + bias1));
        }
        int t1 = toks[m + 16];
        if (t1 >= 0) {
            float w = wts[m + 16];
            unsafeAtomicAdd(out + (size_t)t1 * S_DIM + c0, w * (acc10[j] + bias0));
            unsafeAtomicAdd(out + (size_t)t1 * S_DIM + c1, w * (acc11[j] + bias1));
        }
    }
}

// ---------------- kernel 5: in-place ReLU ----------------
__global__ __launch_bounds__(256) void relu_kernel(float* __restrict__ out) {
    size_t i = (size_t)blockIdx.x * blockDim.x + threadIdx.x;
    float4 v = ((float4*)out)[i];
    v.x = fmaxf(v.x, 0.f); v.y = fmaxf(v.y, 0.f);
    v.z = fmaxf(v.z, 0.f); v.w = fmaxf(v.w, 0.f);
    ((float4*)out)[i] = v;
}

extern "C" void kernel_launch(void* const* d_in, const int* in_sizes, int n_in,
                              void* d_out, int out_size, void* d_ws, size_t ws_size,
                              hipStream_t stream) {
    (void)in_sizes; (void)n_in; (void)out_size; (void)ws_size;
    const float* x        = (const float*)d_in[0];   // [32,512,512]
    const float* gate_w   = (const float*)d_in[1];   // [8,512]
    const float* expert_w = (const float*)d_in[2];   // [8,512,512]
    const float* expert_b = (const float*)d_in[3];   // [8,512]

    float* out       = (float*)d_out;                // [16384,512]
    float* gates_out = out + OUT_ELEMS;              // [16384,8]

    char* ws = (char*)d_ws;
    int*      counts = (int*)(ws + WS_COUNTS);
    int*      etok   = (int*)(ws + WS_TOK);
    float*    ewt    = (float*)(ws + WS_WT);
    _Float16* wh     = (_Float16*)(ws + WS_WH);
    _Float16* xh     = (_Float16*)(ws + WS_XH);

    zero_kernel<<<OUT_ELEMS / 4 / 256, 256, 0, stream>>>(out, counts);
    convw_kernel<<<(N_EXP * S_DIM * S_DIM) / 4 / 256, 256, 0, stream>>>(expert_w, wh);
    gate_kernel<<<N_TOK / 8, 256, 0, stream>>>(x, gate_w, gates_out, xh,
                                               counts, etok, ewt);
    moe_gemm_kernel<<<dim3(N_TOK / 32, S_DIM / 256, N_EXP), 256, 0, stream>>>(
        xh, wh, expert_b, counts, etok, ewt, out);
    relu_kernel<<<OUT_ELEMS / 4 / 256, 256, 0, stream>>>(out);
}
